// AttnBlock_48782238548103
// MI455X (gfx1250) — compile-verified, matrix operand order checked by reading, not run on hardware
//
#include <hip/hip_runtime.h>
#include <hip/hip_fp16.h>

// ---------------------------------------------------------------------------
// AttnBlock for MI455X (gfx1250, wave32, WMMA f16 16x16x32 with f32 accum)
// x:[2,512,64,64] f32. GroupNorm(32) -> QKV proj -> 8-head attn (N=4096,hd=64)
// -> out proj -> residual.
// All GEMMs: v_wmma_f32_16x16x32_f16; tile staging: global_load_async_to_lds.
// V is stored transposed so every WMMA operand loads as two ds_load_b128.
// Q stays resident in LDS (not VGPRs) to keep the flash loop spill-free.
// ---------------------------------------------------------------------------

#define C_DIM  512
#define N_PIX  4096
#define BATCH  2
#define HEADS  8
#define HD     64
#define N_GRP  32

typedef _Float16 half_t;
typedef __attribute__((ext_vector_type(16))) _Float16 v16h;
typedef __attribute__((ext_vector_type(8)))  _Float16 v8h;
typedef __attribute__((ext_vector_type(8)))  float    v8f;
typedef int v4i __attribute__((vector_size(16)));   // matches builtin param type

#if __has_builtin(__builtin_amdgcn_global_load_async_to_lds_b128)
#define HAVE_ASYNC_LDS 1
typedef __attribute__((address_space(1))) v4i* gv4i_p;   // global int4*
typedef __attribute__((address_space(3))) v4i* lv4i_p;   // LDS int4*
#else
#define HAVE_ASYNC_LDS 0
#endif

// Copy 16 bytes global -> LDS. Async (ASYNCcnt-tracked, bypasses VGPRs) when
// the gfx1250 builtin is available; plain vector copy otherwise.
__device__ inline void cp_b128(half_t* l, const half_t* g) {
#if HAVE_ASYNC_LDS
    __builtin_amdgcn_global_load_async_to_lds_b128(
        (gv4i_p)(unsigned long long)g,
        (lv4i_p)(unsigned)(unsigned long long)l,
        0, 0);
#else
    *(v8h*)l = *(const v8h*)g;
#endif
}

__device__ inline void cp_wait() {
#if HAVE_ASYNC_LDS
#if __has_builtin(__builtin_amdgcn_s_wait_asynccnt)
    __builtin_amdgcn_s_wait_asynccnt(0);
#else
    asm volatile("s_wait_asynccnt 0x0" ::: "memory");
#endif
#endif
}

// CDNA5 16-bit A-matrix 16x32 layout (ISA 7.12.2):
// lane L: row r = L%16 ; lanes 0-15 hold K = {0..7,16..23}, lanes 16-31 hold
// K = {8..15,24..31}. Per lane: two contiguous 16B chunks -> 2x ds_load_b128.
__device__ inline v16h load_a16(const half_t* __restrict__ s, int ld) {
    int lane = threadIdx.x & 31;
    int r    = lane & 15;
    int hi   = (lane >> 4) << 3;        // 0 or 8
    const half_t* row = s + r * ld;
    v16h a;
#pragma unroll
    for (int v = 0; v < 8; ++v) {
        int k = ((v & 4) ? 16 : 0) + hi + ((v & 3) << 1);
        a[2 * v]     = row[k];
        a[2 * v + 1] = row[k + 1];
    }
    return a;
}

__device__ inline v8f wmma16(v16h a, v16h b, v8f c) {
    return __builtin_amdgcn_wmma_f32_16x16x32_f16(
        false, a, false, b, (short)0, c, false, false);
}

// ---------------------------------------------------------------------------
// GroupNorm: one block per (batch, group); writes h[b][n][c] as f16
// ---------------------------------------------------------------------------
__global__ void gn_kernel(const float* __restrict__ x,
                          const float* __restrict__ gamma,
                          const float* __restrict__ beta,
                          half_t* __restrict__ h) {
    __shared__ float red[512];
    int b   = blockIdx.x >> 5;
    int g   = blockIdx.x & 31;
    int tid = threadIdx.x;
    const float* xg = x + ((size_t)b * C_DIM + g * 16) * N_PIX;

    float s = 0.f, s2 = 0.f;
    for (int i = tid; i < 16 * N_PIX; i += 256) {
        float v = xg[i];
        s += v; s2 += v * v;
    }
    red[tid] = s; red[256 + tid] = s2;
    __syncthreads();
    for (int off = 128; off > 0; off >>= 1) {
        if (tid < off) {
            red[tid]       += red[tid + off];
            red[256 + tid] += red[256 + tid + off];
        }
        __syncthreads();
    }
    float mean = red[0] * (1.f / 65536.f);
    float var  = red[256] * (1.f / 65536.f) - mean * mean;
    float rstd = rsqrtf(var + 1e-5f);

    for (int i = tid; i < 16 * N_PIX; i += 256) {
        int cl = i & 15, n = i >> 4;
        int c  = g * 16 + cl;
        float v = (xg[cl * N_PIX + n] - mean) * rstd * gamma[c] + beta[c];
        h[((size_t)b * N_PIX + n) * C_DIM + c] = (half_t)v;
    }
}

// ---------------------------------------------------------------------------
// Projection GEMM:  acc = h[m][:] . W[col][:] + bias[col]
// block = 256 thr (8 waves), tile 128(M) x 64(N), K chunks of 32 via LDS.
// TRANS_V=0: dst[b][head][n][d] (q,k).  TRANS_V=1: dst[b][head][d][n] (v^T).
// ---------------------------------------------------------------------------
template <bool TRANS_V>
__global__ void qkv_kernel(const half_t* __restrict__ h,
                           const float* __restrict__ W,
                           const float* __restrict__ bias,
                           half_t* __restrict__ dst) {
    __shared__ alignas(16) half_t ldsA[128 * 32];
    __shared__ alignas(16) half_t ldsB[64 * 32];
    int m0 = blockIdx.x * 128;
    int n0 = blockIdx.y * 64;
    int tid = threadIdx.x, wave = tid >> 5, lane = tid & 31;

    v8f acc[4] = {};
    for (int kc = 0; kc < C_DIM; kc += 32) {
        __syncthreads();
        for (int i = tid; i < 512; i += 256) {           // A: 128x32 f16, async
            int r = i >> 2, c8 = (i & 3) << 3;
            cp_b128(ldsA + r * 32 + c8, h + (size_t)(m0 + r) * C_DIM + kc + c8);
        }
        for (int i = tid; i < 64 * 32; i += 256) {       // B: W f32 -> f16
            int r = i >> 5, c = i & 31;
            ldsB[i] = (half_t)W[(size_t)(n0 + r) * C_DIM + kc + c];
        }
        cp_wait();
        __syncthreads();
        v16h a = load_a16(ldsA + wave * 16 * 32, 32);
#pragma unroll
        for (int nt = 0; nt < 4; ++nt) {
            v16h bm = load_a16(ldsB + nt * 16 * 32, 32);
            acc[nt] = wmma16(a, bm, acc[nt]);
        }
    }
    int hs = lane >> 4, ct = lane & 15;
#pragma unroll
    for (int nt = 0; nt < 4; ++nt) {
        int col  = n0 + nt * 16 + ct;
        float bv = bias[col];
        int head = col >> 6, d = col & 63;
#pragma unroll
        for (int v = 0; v < 8; ++v) {
            int m  = m0 + wave * 16 + v + (hs << 3);
            int bb = m >> 12, n = m & (N_PIX - 1);
            half_t val = (half_t)(acc[nt][v] + bv);
            if (TRANS_V)
                dst[(((size_t)bb * HEADS + head) * HD + d) * N_PIX + n] = val;
            else
                dst[(((size_t)bb * HEADS + head) * N_PIX + n) * HD + d] = val;
        }
    }
}

// ---------------------------------------------------------------------------
// Flash attention: grid (N/128, B*HEADS). Each wave owns 16 query rows.
// q,k: [b][head][n][d] ; vt: [b][head][d][n] ; o: [b][n][c] f16
// Q lives in LDS for the whole loop; fragments are re-read per chunk so the
// softmax/accumulator pressure peak never overlaps a long-lived Q register.
// ---------------------------------------------------------------------------
__global__ void __launch_bounds__(256, 1)
attn_kernel(const half_t* __restrict__ q,
            const half_t* __restrict__ k,
            const half_t* __restrict__ vt,
            half_t* __restrict__ o) {
    __shared__ alignas(16) half_t ldsQ[128 * 64];   // [i][d], resident
    __shared__ alignas(16) half_t ldsK[64 * 64];    // [j][d]
    __shared__ alignas(16) half_t ldsVt[64 * 64];   // [d][j]
    __shared__ alignas(16) half_t ldsP[128 * 64];   // P (per-wave 16x64 slices)
    int m0 = blockIdx.x * 128;
    int bh = blockIdx.y;                            // b*8 + head
    size_t base = (size_t)bh * N_PIX * HD;          // same for q,k,vt
    int tid = threadIdx.x, wave = tid >> 5, lane = tid & 31;
    int hs = lane >> 4, ct = lane & 15;
    const float scale = 0.125f;                     // 1/sqrt(64)

    for (int i = tid; i < 1024; i += 256) {         // stage Q 128x64, async
        int r = i >> 3, c8 = (i & 7) << 3;
        cp_b128(ldsQ + r * 64 + c8, q + base + (size_t)(m0 + r) * HD + c8);
    }
    cp_wait();
    __syncthreads();

    const half_t* myQ = ldsQ + wave * 16 * 64;
    half_t*       myP = ldsP + wave * 16 * 64;

    float mrow[8], lrow[8];
    v8f oacc[4] = {};
#pragma unroll
    for (int r = 0; r < 8; ++r) { mrow[r] = -INFINITY; lrow[r] = 0.f; }

    for (int j0 = 0; j0 < N_PIX; j0 += 64) {
        __syncthreads();
        for (int i = tid; i < 512; i += 256) {      // K tile 64x64 [j][d]
            int r = i >> 3, c8 = (i & 7) << 3;
            cp_b128(ldsK + r * 64 + c8, k + base + (size_t)(j0 + r) * HD + c8);
        }
        for (int i = tid; i < 512; i += 256) {      // V^T tile 64x64 [d][j]
            int r = i >> 3, c8 = (i & 7) << 3;
            cp_b128(ldsVt + r * 64 + c8, vt + base + (size_t)r * N_PIX + j0 + c8);
        }
        cp_wait();
        __syncthreads();

        // S = Q K^T  (Q fragments re-read from LDS -> short live range)
        v16h qreg0 = load_a16(myQ, 64);
        v16h qreg1 = load_a16(myQ + 32, 64);
        v8f s[4];
#pragma unroll
        for (int jt = 0; jt < 4; ++jt) {
            v8f z = {};
            z = wmma16(qreg0, load_a16(ldsK + jt * 16 * 64, 64), z);
            z = wmma16(qreg1, load_a16(ldsK + jt * 16 * 64 + 32, 64), z);
            s[jt] = z;
        }
        // online softmax (row stats live across the 16 lanes of each half)
        float mnew[8];
#pragma unroll
        for (int r = 0; r < 8; ++r) {
            float mx = mrow[r];
#pragma unroll
            for (int jt = 0; jt < 4; ++jt) mx = fmaxf(mx, s[jt][r] * scale);
            for (int d = 1; d < 16; d <<= 1) mx = fmaxf(mx, __shfl_xor(mx, d));
            mnew[r] = mx;
        }
#pragma unroll
        for (int r = 0; r < 8; ++r) {
            float alpha = expf(mrow[r] - mnew[r]);
            float rs = 0.f;
#pragma unroll
            for (int jt = 0; jt < 4; ++jt) {
                float p = expf(s[jt][r] * scale - mnew[r]);
                s[jt][r] = p;
                rs += p;
            }
            for (int d = 1; d < 16; d <<= 1) rs += __shfl_xor(rs, d);
            lrow[r] = lrow[r] * alpha + rs;
            mrow[r] = mnew[r];
#pragma unroll
            for (int dt = 0; dt < 4; ++dt) oacc[dt][r] *= alpha;
        }
        // C-layout -> A-layout via per-wave LDS scratch
#pragma unroll
        for (int jt = 0; jt < 4; ++jt)
#pragma unroll
            for (int r = 0; r < 8; ++r)
                myP[(r + (hs << 3)) * 64 + jt * 16 + ct] = (half_t)s[jt][r];
        __syncthreads();
        v16h a0 = load_a16(myP, 64);                // j = 0..31
        v16h a1 = load_a16(myP + 32, 64);           // j = 32..63
#pragma unroll
        for (int dt = 0; dt < 4; ++dt) {            // O += P V
            v16h b0 = load_a16(ldsVt + dt * 16 * 64, 64);        // j 0..31
            v16h b1 = load_a16(ldsVt + dt * 16 * 64 + 32, 64);   // j 32..63
            oacc[dt] = wmma16(a0, b0, oacc[dt]);
            oacc[dt] = wmma16(a1, b1, oacc[dt]);
        }
    }
    int head = bh & 7, bb = bh >> 3;
#pragma unroll
    for (int dt = 0; dt < 4; ++dt)
#pragma unroll
        for (int r = 0; r < 8; ++r) {
            int m = m0 + wave * 16 + r + (hs << 3);
            int c = head * 64 + dt * 16 + ct;
            o[((size_t)bb * N_PIX + m) * C_DIM + c] =
                (half_t)(oacc[dt][r] / lrow[r]);
        }
}

// ---------------------------------------------------------------------------
// Output projection + residual:  out[b][c][n] = x[b][c][n] + o.Wo^T + bo
// ---------------------------------------------------------------------------
__global__ void proj_kernel(const half_t* __restrict__ ob,
                            const float* __restrict__ W,
                            const float* __restrict__ bias,
                            const float* __restrict__ x,
                            float* __restrict__ out) {
    __shared__ alignas(16) half_t ldsA[128 * 32];
    __shared__ alignas(16) half_t ldsB[64 * 32];
    int m0 = blockIdx.x * 128;
    int n0 = blockIdx.y * 64;
    int tid = threadIdx.x, wave = tid >> 5, lane = tid & 31;

    v8f acc[4] = {};
    for (int kc = 0; kc < C_DIM; kc += 32) {
        __syncthreads();
        for (int i = tid; i < 512; i += 256) {
            int r = i >> 2, c8 = (i & 3) << 3;
            cp_b128(ldsA + r * 32 + c8, ob + (size_t)(m0 + r) * C_DIM + kc + c8);
        }
        for (int i = tid; i < 64 * 32; i += 256) {
            int r = i >> 5, c = i & 31;
            ldsB[i] = (half_t)W[(size_t)(n0 + r) * C_DIM + kc + c];
        }
        cp_wait();
        __syncthreads();
        v16h a = load_a16(ldsA + wave * 16 * 32, 32);
#pragma unroll
        for (int nt = 0; nt < 4; ++nt) {
            v16h bm = load_a16(ldsB + nt * 16 * 32, 32);
            acc[nt] = wmma16(a, bm, acc[nt]);
        }
    }
    int hs = lane >> 4, ct = lane & 15;
#pragma unroll
    for (int nt = 0; nt < 4; ++nt) {
        int col  = n0 + nt * 16 + ct;
        float bv = bias[col];
#pragma unroll
        for (int v = 0; v < 8; ++v) {
            int m  = m0 + wave * 16 + v + (hs << 3);
            int bb = m >> 12, n = m & (N_PIX - 1);
            size_t idx = ((size_t)bb * C_DIM + col) * N_PIX + n;
            out[idx] = x[idx] + acc[nt][v] + bv;
        }
    }
}

// ---------------------------------------------------------------------------
extern "C" void kernel_launch(void* const* d_in, const int* in_sizes, int n_in,
                              void* d_out, int out_size, void* d_ws, size_t ws_size,
                              hipStream_t stream) {
    const float* x     = (const float*)d_in[0];
    const float* gamma = (const float*)d_in[1];
    const float* beta  = (const float*)d_in[2];
    const float* Wq = (const float*)d_in[3];
    const float* bq = (const float*)d_in[4];
    const float* Wk = (const float*)d_in[5];
    const float* bk = (const float*)d_in[6];
    const float* Wv = (const float*)d_in[7];
    const float* bv = (const float*)d_in[8];
    const float* Wo = (const float*)d_in[9];
    const float* bo = (const float*)d_in[10];
    float* out = (float*)d_out;

    size_t sz = (size_t)BATCH * N_PIX * C_DIM;      // 4M halves per buffer
    half_t* h    = (half_t*)d_ws;
    half_t* qbuf = h + sz;
    half_t* kbuf = qbuf + sz;
    half_t* vbuf = kbuf + sz;
    half_t* obuf = h;                               // h dead after QKV -> reuse

    gn_kernel<<<dim3(BATCH * N_GRP), dim3(256), 0, stream>>>(x, gamma, beta, h);
    qkv_kernel<false><<<dim3(64, 8), dim3(256), 0, stream>>>(h, Wq, bq, qbuf);
    qkv_kernel<false><<<dim3(64, 8), dim3(256), 0, stream>>>(h, Wk, bk, kbuf);
    qkv_kernel<true ><<<dim3(64, 8), dim3(256), 0, stream>>>(h, Wv, bv, vbuf);
    attn_kernel<<<dim3(32, 16), dim3(256), 0, stream>>>(qbuf, kbuf, vbuf, obuf);
    proj_kernel<<<dim3(64, 8), dim3(256), 0, stream>>>(obuf, Wo, bo, x, out);
}
